// AttentionLayer_67817533604501
// MI455X (gfx1250) — matrix operand order repaired and checked by reading, compiled, not run on hardware
//
#include <hip/hip_runtime.h>

// ---------------------------------------------------------------------------
// Gated multi-head attention for MI455X (gfx1250), wave32 + WMMA bf16.
// B=4, L=S=1024, D_MODEL=1024, H=16, DK=64.
//
// Pipeline:
//   1) proj_gemm  x3 : Q/K/V = X @ (gate[b]*W)^T + bias   (V stored transposed)
//   2) score_gemm    : attn_raw = (Q @ K^T) * 1/sqrt(DK)  -> d_out attn region
//   3) softmax_rows  : softmax over S, in place
//   4) av_gemm       : ctx = attn @ V                      (bf16 ctx in ws)
//   5) out_gemm      : out = ctx @ W_o^T + b_o             -> d_out out region
//
// GEMM core: 128x64 block tile, 8 waves each owning a 2x2 grid of 16x16
// accumulators (4 independent WMMAs per k-chunk, 2 LDS b128 loads per WMMA).
// K-stage 64, double-buffered LDS, ONE barrier per stage; next-stage global
// loads are issued before the WMMAs so VMEM latency hides under matrix math.
// bf16-source tiles are staged with GLOBAL_LOAD_ASYNC_TO_LDS_B128 (ASYNCcnt).
// ---------------------------------------------------------------------------

typedef __bf16 bf16_t;
typedef __attribute__((ext_vector_type(16))) __bf16 v16bf;
typedef __attribute__((ext_vector_type(8)))  __bf16 v8bf;
typedef __attribute__((ext_vector_type(8)))  float   v8f;
typedef __attribute__((ext_vector_type(4)))  int     v4i;

#define DM     1024        // d_model == H*DK
#define TM     128         // block tile M
#define TN     64          // block tile N (== DK)
#define KT     64          // K elements per LDS stage (2 WMMA k-chunks)
#define LDSS   72          // LDS row stride in bf16 (144B: 16B-aligned, bank-friendly)
#define ABUF   (TM * LDSS)
#define BBUF   (TN * LDSS)

// ---------------- bf16 conversion helpers ----------------------------------
static __device__ __forceinline__ bf16_t f2bf(float f) {
  unsigned u = __builtin_bit_cast(unsigned, f);
  u += 0x7fffu + ((u >> 16) & 1u);               // round-to-nearest-even
  unsigned short s = (unsigned short)(u >> 16);
  return __builtin_bit_cast(bf16_t, s);
}

static __device__ __forceinline__ v8bf pack8(float4 a, float4 b) {
  v8bf v;
  v[0] = f2bf(a.x); v[1] = f2bf(a.y); v[2] = f2bf(a.z); v[3] = f2bf(a.w);
  v[4] = f2bf(b.x); v[5] = f2bf(b.y); v[6] = f2bf(b.z); v[7] = f2bf(b.w);
  return v;
}

static __device__ __forceinline__ float4 mul4(float4 a, float4 b) {
  return make_float4(a.x * b.x, a.y * b.y, a.z * b.z, a.w * b.w);
}

static __device__ __forceinline__ v16bf cat16(v8bf lo, v8bf hi) {
  return __builtin_shufflevector(lo, hi, 0,1,2,3,4,5,6,7,8,9,10,11,12,13,14,15);
}

// ---------------- async global->LDS (CDNA5, ASYNCcnt) -----------------------
#if defined(__has_builtin)
#if __has_builtin(__builtin_amdgcn_global_load_async_to_lds_b128)
#define HAVE_ASYNC_LDS 1
#endif
#endif

typedef __attribute__((address_space(1))) v4i gas_v4i;   // global
typedef __attribute__((address_space(3))) v4i las_v4i;   // LDS

static __device__ __forceinline__ void async_copy16(const bf16_t* g, bf16_t* l) {
#if defined(HAVE_ASYNC_LDS)
  __builtin_amdgcn_global_load_async_to_lds_b128(
      (gas_v4i*)(g), (las_v4i*)(l), 0, 0);
#else
  *(uint4*)l = *(const uint4*)g;
#endif
}

static __device__ __forceinline__ void async_wait() {
#if defined(HAVE_ASYNC_LDS)
#if __has_builtin(__builtin_amdgcn_s_wait_asynccnt)
  __builtin_amdgcn_s_wait_asynccnt(0);
#else
  asm volatile("s_wait_asynccnt 0x0" ::: "memory");
#endif
#endif
}

// ---------------- WMMA fragments (ISA 7.12.2 layouts) -----------------------
// A fragment, 16x32 bf16: lane l -> row l&15; K halves selected by l>>4.
static __device__ __forceinline__ v16bf load_frag_a(const bf16_t* s, int mi, int lane) {
  const int m  = mi * 16 + (lane & 15);
  const int kh = (lane >> 4) & 1;
  v8bf lo = *(const v8bf*)(s + m * LDSS + kh * 8);
  v8bf hi = *(const v8bf*)(s + m * LDSS + 16 + kh * 8);
  return cat16(lo, hi);
}

// B fragment, 32x16 bf16: lane l -> col l&15; K halves selected by l>>4 (same as A).
// LDS B tile is stored [n][k].
static __device__ __forceinline__ v16bf load_frag_b(const bf16_t* s, int nj, int lane) {
  const int n  = nj * 16 + (lane & 15);
  const int kh = (lane >> 4) & 1;
  v8bf lo = *(const v8bf*)(s + n * LDSS + kh * 8);
  v8bf hi = *(const v8bf*)(s + n * LDSS + 16 + kh * 8);
  return cat16(lo, hi);
}

static __device__ __forceinline__ v8f wmma_bf16(v16bf a, v16bf b, v8f c) {
  return __builtin_amdgcn_wmma_f32_16x16x32_bf16(
      /*neg_a=*/false, a, /*neg_b=*/false, b,
      /*c_mod=*/(short)0, c, /*reuse_a=*/false, /*reuse_b=*/false);
}

// One 64-wide K stage: 2 k-chunks x (2 A-frags, 2 B-frags, 4 independent WMMAs).
static __device__ __forceinline__ void mma_stage4(
    const bf16_t* As_, const bf16_t* Bs_, int mi0, int nj0, int lane, v8f* acc)
{
#pragma unroll
  for (int kk = 0; kk < KT; kk += 32) {
    const v16bf a0 = load_frag_a(As_ + kk, mi0,     lane);
    const v16bf a1 = load_frag_a(As_ + kk, mi0 + 1, lane);
    const v16bf b0 = load_frag_b(Bs_ + kk, nj0,     lane);
    const v16bf b1 = load_frag_b(Bs_ + kk, nj0 + 1, lane);
    acc[0] = wmma_bf16(a0, b0, acc[0]);
    acc[1] = wmma_bf16(a0, b1, acc[1]);
    acc[2] = wmma_bf16(a1, b0, acc[2]);
    acc[3] = wmma_bf16(a1, b1, acc[3]);
  }
}

// ---------------------------------------------------------------------------
// 1) Gated projection: Y[g, n] = sum_k X[g, k] * (gate[b, n, k] * W[n, k]) + bias[n]
//    g = b*L + l over [0, 4096). If vtranspose: store Y as [b][h][dk][s] bf16.
// ---------------------------------------------------------------------------
__global__ __launch_bounds__(256) void proj_gemm_kernel(
    const float* __restrict__ X, const float* __restrict__ W,
    const float* __restrict__ gate, const float* __restrict__ bias,
    bf16_t* __restrict__ Y, int vtranspose)
{
  __shared__ bf16_t As[2][ABUF];
  __shared__ bf16_t Bs[2][BBUF];

  const int t    = threadIdx.x;
  const int lane = t & 31;
  const int wave = t >> 5;
  const int m0   = blockIdx.x * TM;            // global row in [0, 4096)
  const int n0   = blockIdx.y * TN;            // output feature
  const int b    = m0 >> 10;                   // batch index (TM divides L)
  const int mi0  = (wave >> 1) * 2;            // 0,2,4,6
  const int nj0  = (wave & 1) * 2;             // 0 or 2

  const int srow = t >> 2;                     // 0..63 staging row
  const int scol = (t & 3) * 16;               // 0,16,32,48 staging col
  const int soff  = srow * LDSS + scol;
  const int soff2 = soff + 64 * LDSS;          // second A half (rows 64..127)

  const float* xrow0 = X + (size_t)(m0 + srow) * DM + scol;
  const float* xrow1 = xrow0 + (size_t)64 * DM;
  const float* wrow  = W    + (size_t)(n0 + srow) * DM + scol;
  const float* grow  = gate + ((size_t)b * DM + (n0 + srow)) * DM + scol;

  v8f acc[4] = {};

  // ---- prologue: stage chunk 0 into buffer 0 ----
  {
    const float4 a0 = *(const float4*)(xrow0);      const float4 a1 = *(const float4*)(xrow0 + 4);
    const float4 a2 = *(const float4*)(xrow0 + 8);  const float4 a3 = *(const float4*)(xrow0 + 12);
    const float4 a4 = *(const float4*)(xrow1);      const float4 a5 = *(const float4*)(xrow1 + 4);
    const float4 a6 = *(const float4*)(xrow1 + 8);  const float4 a7 = *(const float4*)(xrow1 + 12);
    const float4 w0 = mul4(*(const float4*)(wrow),      *(const float4*)(grow));
    const float4 w1 = mul4(*(const float4*)(wrow + 4),  *(const float4*)(grow + 4));
    const float4 w2 = mul4(*(const float4*)(wrow + 8),  *(const float4*)(grow + 8));
    const float4 w3 = mul4(*(const float4*)(wrow + 12), *(const float4*)(grow + 12));
    *(v8bf*)&As[0][soff]      = pack8(a0, a1);
    *(v8bf*)&As[0][soff + 8]  = pack8(a2, a3);
    *(v8bf*)&As[0][soff2]     = pack8(a4, a5);
    *(v8bf*)&As[0][soff2 + 8] = pack8(a6, a7);
    *(v8bf*)&Bs[0][soff]      = pack8(w0, w1);
    *(v8bf*)&Bs[0][soff + 8]  = pack8(w2, w3);
  }
  __syncthreads();

  int cur = 0;
  for (int kc = KT; kc < DM; kc += KT) {
    if (kc + KT < DM) {                        // global_prefetch_b8 for stage+2
      __builtin_prefetch(xrow0 + kc + KT, 0, 0);
      __builtin_prefetch(xrow1 + kc + KT, 0, 0);
      __builtin_prefetch(wrow + kc + KT, 0, 0);
      __builtin_prefetch(grow + kc + KT, 0, 0);
    }
    // next-stage global loads (in flight during WMMAs below); gate applied at load
    const float4 a0 = *(const float4*)(xrow0 + kc);      const float4 a1 = *(const float4*)(xrow0 + kc + 4);
    const float4 a2 = *(const float4*)(xrow0 + kc + 8);  const float4 a3 = *(const float4*)(xrow0 + kc + 12);
    const float4 a4 = *(const float4*)(xrow1 + kc);      const float4 a5 = *(const float4*)(xrow1 + kc + 4);
    const float4 a6 = *(const float4*)(xrow1 + kc + 8);  const float4 a7 = *(const float4*)(xrow1 + kc + 12);
    const float4 w0 = mul4(*(const float4*)(wrow + kc),      *(const float4*)(grow + kc));
    const float4 w1 = mul4(*(const float4*)(wrow + kc + 4),  *(const float4*)(grow + kc + 4));
    const float4 w2 = mul4(*(const float4*)(wrow + kc + 8),  *(const float4*)(grow + kc + 8));
    const float4 w3 = mul4(*(const float4*)(wrow + kc + 12), *(const float4*)(grow + kc + 12));

    mma_stage4(As[cur], Bs[cur], mi0, nj0, lane, acc);

    const int nxt = cur ^ 1;
    *(v8bf*)&As[nxt][soff]      = pack8(a0, a1);
    *(v8bf*)&As[nxt][soff + 8]  = pack8(a2, a3);
    *(v8bf*)&As[nxt][soff2]     = pack8(a4, a5);
    *(v8bf*)&As[nxt][soff2 + 8] = pack8(a6, a7);
    *(v8bf*)&Bs[nxt][soff]      = pack8(w0, w1);
    *(v8bf*)&Bs[nxt][soff + 8]  = pack8(w2, w3);
    __syncthreads();
    cur = nxt;
  }
  mma_stage4(As[cur], Bs[cur], mi0, nj0, lane, acc);

  const int ln = lane & 15;
  const int lm = (lane >> 4) * 8;
#pragma unroll
  for (int ms = 0; ms < 2; ++ms) {
#pragma unroll
    for (int ns = 0; ns < 2; ++ns) {
      const v8f a = acc[ms * 2 + ns];
      const int col = n0 + (nj0 + ns) * 16 + ln;
      const float bb = bias[col];
      for (int j = 0; j < 8; ++j) {
        const int gr = m0 + (mi0 + ms) * 16 + lm + j;
        const float v = a[j] + bb;
        if (!vtranspose) {
          Y[(size_t)gr * DM + col] = f2bf(v);
        } else {
          const int bb2 = gr >> 10, s = gr & 1023;
          const int h = col >> 6, dk = col & 63;
          Y[(((size_t)bb2 * 16 + h) * 64 + dk) * 1024 + s] = f2bf(v);
        }
      }
    }
  }
}

// ---------------------------------------------------------------------------
// 2) Scores: attn_raw[b,h,l,s] = (Q[b,l,h,:] . K[b,s,h,:]) * 0.125   (DK=64)
//    Single K stage; Q/K tiles staged with async global->LDS copies.
// ---------------------------------------------------------------------------
__global__ __launch_bounds__(256) void score_gemm_kernel(
    const bf16_t* __restrict__ Qp, const bf16_t* __restrict__ Kp,
    float* __restrict__ attn)
{
  __shared__ bf16_t As[ABUF];
  __shared__ bf16_t Bs[BBUF];

  const int t = threadIdx.x, lane = t & 31, wave = t >> 5;
  const int bz = blockIdx.z;                   // b*16 + h
  const int b = bz >> 4, h = bz & 15;
  const int l0 = blockIdx.x * TM;
  const int s0 = blockIdx.y * TN;
  const int mi0 = (wave >> 1) * 2, nj0 = (wave & 1) * 2;
  const int srow = t >> 2, scol = (t & 3) * 16;
  const int soff  = srow * LDSS + scol;
  const int soff2 = soff + 64 * LDSS;

  const bf16_t* qrow0 = Qp + (size_t)(b * 1024 + l0 + srow) * DM + h * 64 + scol;
  const bf16_t* qrow1 = qrow0 + (size_t)64 * DM;
  const bf16_t* krow  = Kp + (size_t)(b * 1024 + s0 + srow) * DM + h * 64 + scol;

  async_copy16(qrow0,     &As[soff]);
  async_copy16(qrow0 + 8, &As[soff + 8]);
  async_copy16(qrow1,     &As[soff2]);
  async_copy16(qrow1 + 8, &As[soff2 + 8]);
  async_copy16(krow,      &Bs[soff]);
  async_copy16(krow + 8,  &Bs[soff + 8]);
  async_wait();
  __syncthreads();

  v8f acc[4] = {};
  mma_stage4(As, Bs, mi0, nj0, lane, acc);

  const int ln = lane & 15, lm = (lane >> 4) * 8;
#pragma unroll
  for (int ms = 0; ms < 2; ++ms) {
#pragma unroll
    for (int ns = 0; ns < 2; ++ns) {
      const v8f a = acc[ms * 2 + ns];
      const int col = s0 + (nj0 + ns) * 16 + ln;
      for (int j = 0; j < 8; ++j) {
        const int row = l0 + (mi0 + ms) * 16 + lm + j;
        attn[((size_t)bz * 1024 + row) * 1024 + col] = a[j] * 0.125f;
      }
    }
  }
}

// ---------------------------------------------------------------------------
// 3) Row softmax over S=1024, in place. One block per (b,h,l) row.
// ---------------------------------------------------------------------------
__global__ __launch_bounds__(256) void softmax_rows_kernel(float* __restrict__ attn)
{
  __shared__ float red[256];
  const int t = threadIdx.x;
  float* row = attn + (size_t)blockIdx.x * 1024;

  float x[4];
  float mx = -3.4e38f;
  for (int i = 0; i < 4; ++i) { x[i] = row[t + i * 256]; mx = fmaxf(mx, x[i]); }
  red[t] = mx; __syncthreads();
  for (int s = 128; s > 0; s >>= 1) {
    if (t < s) red[t] = fmaxf(red[t], red[t + s]);
    __syncthreads();
  }
  const float m = red[0]; __syncthreads();

  float sum = 0.f;
  for (int i = 0; i < 4; ++i) { x[i] = __expf(x[i] - m); sum += x[i]; }
  red[t] = sum; __syncthreads();
  for (int s = 128; s > 0; s >>= 1) {
    if (t < s) red[t] += red[t + s];
    __syncthreads();
  }
  const float inv = 1.f / red[0];
  for (int i = 0; i < 4; ++i) row[t + i * 256] = x[i] * inv;
}

// ---------------------------------------------------------------------------
// 4) ctx[b,l,h*64+dk] = sum_s attn[b,h,l,s] * V[b,s,h,dk]   (V stored [b,h,dk,s])
//    A (attn, f32) staged via registers; B (Vt, bf16) staged async.
// ---------------------------------------------------------------------------
__global__ __launch_bounds__(256) void av_gemm_kernel(
    const float* __restrict__ attn, const bf16_t* __restrict__ Vt,
    bf16_t* __restrict__ ctx)
{
  __shared__ bf16_t As[2][ABUF];
  __shared__ bf16_t Bs[2][BBUF];

  const int t = threadIdx.x, lane = t & 31, wave = t >> 5;
  const int bz = blockIdx.z;                   // b*16 + h
  const int b = bz >> 4, h = bz & 15;
  const int l0 = blockIdx.x * TM;              // n0 = 0 (N == DK == 64)
  const int mi0 = (wave >> 1) * 2, nj0 = (wave & 1) * 2;
  const int srow = t >> 2, scol = (t & 3) * 16;
  const int soff  = srow * LDSS + scol;
  const int soff2 = soff + 64 * LDSS;

  const float*  arow0 = attn + ((size_t)bz * 1024 + l0 + srow) * 1024 + scol;
  const float*  arow1 = arow0 + (size_t)64 * 1024;
  const bf16_t* vrow  = Vt + ((size_t)bz * 64 + srow) * 1024 + scol;

  v8f acc[4] = {};

  // ---- prologue ----
  {
    async_copy16(vrow,     &Bs[0][soff]);
    async_copy16(vrow + 8, &Bs[0][soff + 8]);
    const float4 a0 = *(const float4*)(arow0);      const float4 a1 = *(const float4*)(arow0 + 4);
    const float4 a2 = *(const float4*)(arow0 + 8);  const float4 a3 = *(const float4*)(arow0 + 12);
    const float4 a4 = *(const float4*)(arow1);      const float4 a5 = *(const float4*)(arow1 + 4);
    const float4 a6 = *(const float4*)(arow1 + 8);  const float4 a7 = *(const float4*)(arow1 + 12);
    *(v8bf*)&As[0][soff]      = pack8(a0, a1);
    *(v8bf*)&As[0][soff + 8]  = pack8(a2, a3);
    *(v8bf*)&As[0][soff2]     = pack8(a4, a5);
    *(v8bf*)&As[0][soff2 + 8] = pack8(a6, a7);
    async_wait();
  }
  __syncthreads();

  int cur = 0;
  for (int kc = KT; kc < 1024; kc += KT) {
    const int nxt = cur ^ 1;
    // next-stage traffic in flight during WMMAs
    async_copy16(vrow + kc,     &Bs[nxt][soff]);
    async_copy16(vrow + kc + 8, &Bs[nxt][soff + 8]);
    const float4 a0 = *(const float4*)(arow0 + kc);      const float4 a1 = *(const float4*)(arow0 + kc + 4);
    const float4 a2 = *(const float4*)(arow0 + kc + 8);  const float4 a3 = *(const float4*)(arow0 + kc + 12);
    const float4 a4 = *(const float4*)(arow1 + kc);      const float4 a5 = *(const float4*)(arow1 + kc + 4);
    const float4 a6 = *(const float4*)(arow1 + kc + 8);  const float4 a7 = *(const float4*)(arow1 + kc + 12);

    mma_stage4(As[cur], Bs[cur], mi0, nj0, lane, acc);

    *(v8bf*)&As[nxt][soff]      = pack8(a0, a1);
    *(v8bf*)&As[nxt][soff + 8]  = pack8(a2, a3);
    *(v8bf*)&As[nxt][soff2]     = pack8(a4, a5);
    *(v8bf*)&As[nxt][soff2 + 8] = pack8(a6, a7);
    async_wait();
    __syncthreads();
    cur = nxt;
  }
  mma_stage4(As[cur], Bs[cur], mi0, nj0, lane, acc);

  const int ln = lane & 15, lm = (lane >> 4) * 8;
#pragma unroll
  for (int ms = 0; ms < 2; ++ms) {
#pragma unroll
    for (int ns = 0; ns < 2; ++ns) {
      const v8f a = acc[ms * 2 + ns];
      const int col = (nj0 + ns) * 16 + ln;    // dk in [0,64)
      for (int j = 0; j < 8; ++j) {
        const int l = l0 + (mi0 + ms) * 16 + lm + j;
        ctx[((size_t)(b * 1024 + l)) * DM + h * 64 + col] = f2bf(a[j]);
      }
    }
  }
}

// ---------------------------------------------------------------------------
// 5) out[g, n] = sum_k ctx[g, k] * W_o[n, k] + b_o[n],  g over [0, 4096)
//    A (ctx, bf16) staged async; B (W_o, f32) staged via registers.
// ---------------------------------------------------------------------------
__global__ __launch_bounds__(256) void out_gemm_kernel(
    const bf16_t* __restrict__ ctx, const float* __restrict__ Wo,
    const float* __restrict__ bo, float* __restrict__ out)
{
  __shared__ bf16_t As[2][ABUF];
  __shared__ bf16_t Bs[2][BBUF];

  const int t = threadIdx.x, lane = t & 31, wave = t >> 5;
  const int m0 = blockIdx.x * TM;
  const int n0 = blockIdx.y * TN;
  const int mi0 = (wave >> 1) * 2, nj0 = (wave & 1) * 2;
  const int srow = t >> 2, scol = (t & 3) * 16;
  const int soff  = srow * LDSS + scol;
  const int soff2 = soff + 64 * LDSS;

  const bf16_t* crow0 = ctx + (size_t)(m0 + srow) * DM + scol;
  const bf16_t* crow1 = crow0 + (size_t)64 * DM;
  const float*  wrow  = Wo + (size_t)(n0 + srow) * DM + scol;

  v8f acc[4] = {};

  // ---- prologue ----
  {
    async_copy16(crow0,     &As[0][soff]);
    async_copy16(crow0 + 8, &As[0][soff + 8]);
    async_copy16(crow1,     &As[0][soff2]);
    async_copy16(crow1 + 8, &As[0][soff2 + 8]);
    const float4 w0 = *(const float4*)(wrow);      const float4 w1 = *(const float4*)(wrow + 4);
    const float4 w2 = *(const float4*)(wrow + 8);  const float4 w3 = *(const float4*)(wrow + 12);
    *(v8bf*)&Bs[0][soff]     = pack8(w0, w1);
    *(v8bf*)&Bs[0][soff + 8] = pack8(w2, w3);
    async_wait();
  }
  __syncthreads();

  int cur = 0;
  for (int kc = KT; kc < DM; kc += KT) {
    const int nxt = cur ^ 1;
    async_copy16(crow0 + kc,     &As[nxt][soff]);
    async_copy16(crow0 + kc + 8, &As[nxt][soff + 8]);
    async_copy16(crow1 + kc,     &As[nxt][soff2]);
    async_copy16(crow1 + kc + 8, &As[nxt][soff2 + 8]);
    const float4 w0 = *(const float4*)(wrow + kc);      const float4 w1 = *(const float4*)(wrow + kc + 4);
    const float4 w2 = *(const float4*)(wrow + kc + 8);  const float4 w3 = *(const float4*)(wrow + kc + 12);

    mma_stage4(As[cur], Bs[cur], mi0, nj0, lane, acc);

    *(v8bf*)&Bs[nxt][soff]     = pack8(w0, w1);
    *(v8bf*)&Bs[nxt][soff + 8] = pack8(w2, w3);
    async_wait();
    __syncthreads();
    cur = nxt;
  }
  mma_stage4(As[cur], Bs[cur], mi0, nj0, lane, acc);

  const int ln = lane & 15, lm = (lane >> 4) * 8;
#pragma unroll
  for (int ms = 0; ms < 2; ++ms) {
#pragma unroll
    for (int ns = 0; ns < 2; ++ns) {
      const v8f a = acc[ms * 2 + ns];
      const int col = n0 + (nj0 + ns) * 16 + ln;
      const float bb = bo[col];
      for (int j = 0; j < 8; ++j) {
        const int row = m0 + (mi0 + ms) * 16 + lm + j;
        out[(size_t)row * DM + col] = a[j] + bb;
      }
    }
  }
}

// ---------------------------------------------------------------------------
extern "C" void kernel_launch(void* const* d_in, const int* in_sizes, int n_in,
                              void* d_out, int out_size, void* d_ws, size_t ws_size,
                              hipStream_t stream) {
  (void)in_sizes; (void)n_in; (void)out_size; (void)ws_size;

  const float* queries = (const float*)d_in[0];
  const float* keys    = (const float*)d_in[1];
  const float* values  = (const float*)d_in[2];
  const float* gate    = (const float*)d_in[3];
  const float* W_q     = (const float*)d_in[4];
  const float* b_q     = (const float*)d_in[5];
  const float* W_k     = (const float*)d_in[6];
  const float* b_k     = (const float*)d_in[7];
  const float* W_v     = (const float*)d_in[8];
  const float* b_v     = (const float*)d_in[9];
  const float* W_o     = (const float*)d_in[10];
  const float* b_o     = (const float*)d_in[11];

  const size_t MAT = (size_t)4096 * 1024;      // B*L x D_MODEL elements
  bf16_t* Qp  = (bf16_t*)d_ws;
  bf16_t* Kp  = Qp + MAT;
  bf16_t* Vt  = Kp + MAT;                      // [b][h][dk][s]
  bf16_t* ctx = Vt + MAT;

  float* out  = (float*)d_out;                 // [B, L, H*DK]
  float* attn = out + MAT;                     // [B, H, L, S]

  const dim3 blk(256);

  proj_gemm_kernel<<<dim3(32, 16), blk, 0, stream>>>(queries, W_q, gate, b_q, Qp, 0);
  proj_gemm_kernel<<<dim3(32, 16), blk, 0, stream>>>(keys,    W_k, gate, b_k, Kp, 0);
  proj_gemm_kernel<<<dim3(32, 16), blk, 0, stream>>>(values,  W_v, gate, b_v, Vt, 1);

  score_gemm_kernel<<<dim3(8, 16, 64), blk, 0, stream>>>(Qp, Kp, attn);
  softmax_rows_kernel<<<dim3(65536), blk, 0, stream>>>(attn);
  av_gemm_kernel<<<dim3(8, 1, 64), blk, 0, stream>>>(attn, Vt, ctx);
  out_gemm_kernel<<<dim3(32, 16), blk, 0, stream>>>(ctx, W_o, b_o, out);
}
